// NeuralHMM_8186207666609
// MI455X (gfx1250) — compile-verified, matrix operand order checked by reading, not run on hardware
//
#include <hip/hip_runtime.h>
#include <hip/hip_bf16.h>
#include <math.h>

#define Bsz   128
#define Tlen  512
#define Dd    768
#define Hh    16
#define Ncols 512
#define Mrows (Bsz*Tlen)
#define LDT   40   // padded LDS stride (bf16 elems) -> conflict-free b128 reads

typedef __attribute__((ext_vector_type(16))) __bf16 v16bf;
typedef __attribute__((ext_vector_type(8)))  float  v8f;

union Frag { uint4 q[2]; v16bf v; };

__device__ __forceinline__ unsigned short f2bf(float f){
  unsigned int u = __float_as_uint(f);
  u += 0x7fffu + ((u >> 16) & 1u);          // round-to-nearest-even
  return (unsigned short)(u >> 16);
}

// ---------------------------------------------------------------------------
// Kernel 1: C(65536x512) = emb(65536x768) @ [W_trans | W_emiss](768x512)
// bf16 WMMA, f32 accumulate. Block = 128 thr (4 waves), tile 64x64, K-step 32.
// ---------------------------------------------------------------------------
__global__ __launch_bounds__(128) void gemm_bf16(
    const float* __restrict__ emb, const float* __restrict__ Wt,
    const float* __restrict__ We, float* __restrict__ logits){
  __shared__ unsigned short As[64*LDT];
  __shared__ unsigned short Bs[64*LDT];
  const int tid   = threadIdx.x;
  const int mBase = blockIdx.x * 64;
  const int nBlk  = blockIdx.y;                  // 0..7 (0-3 trans, 4-7 emiss)
  const float* W  = (nBlk < 4) ? Wt : We;
  const int wcol  = (nBlk & 3) * 64;
  const int w = tid >> 5, lane = tid & 31;
  const int ml = lane & 15, khalf = lane >> 4;

  v8f acc[4];
  #pragma unroll
  for (int nt = 0; nt < 4; ++nt)
    #pragma unroll
    for (int e = 0; e < 8; ++e) acc[nt][e] = 0.f;

  for (int k0 = 0; k0 < Dd; k0 += 32){
    if (k0 + 32 < Dd){  // prefetch next K-tile (global_prefetch_b8)
      __builtin_prefetch(emb + (size_t)(mBase + (tid & 63))*Dd + k0 + 32, 0, 1);
      __builtin_prefetch(W   + (size_t)(k0 + 32 + (tid & 31))*256 + wcol, 0, 1);
    }
    // A tile: 64 rows x 32 k, fp32 -> bf16, LDS [m][k]
    #pragma unroll
    for (int q = 0; q < 4; ++q){
      int li = tid + q*128;
      int row = li >> 3, kq = li & 7;
      float4 f = *(const float4*)(emb + (size_t)(mBase + row)*Dd + k0 + kq*4);
      unsigned short* p = As + row*LDT + kq*4;
      p[0]=f2bf(f.x); p[1]=f2bf(f.y); p[2]=f2bf(f.z); p[3]=f2bf(f.w);
    }
    // B tile: 32 k x 64 n, stored transposed LDS [n][k]
    #pragma unroll
    for (int q = 0; q < 4; ++q){
      int li = tid + q*128;
      int kk = li >> 4, nq = li & 15;
      float4 f = *(const float4*)(W + (size_t)(k0 + kk)*256 + wcol + nq*4);
      Bs[(nq*4+0)*LDT + kk] = f2bf(f.x);
      Bs[(nq*4+1)*LDT + kk] = f2bf(f.y);
      Bs[(nq*4+2)*LDT + kk] = f2bf(f.z);
      Bs[(nq*4+3)*LDT + kk] = f2bf(f.w);
    }
    __syncthreads();
    // A fragment: lane (0-15)=M row, lane>>4 selects K-halves {0-7,16-23} vs {8-15,24-31}
    Frag a;
    const int mrow = w*16 + ml;
    a.q[0] = *(const uint4*)(As + mrow*LDT +      khalf*8);
    a.q[1] = *(const uint4*)(As + mrow*LDT + 16 + khalf*8);
    #pragma unroll
    for (int nt = 0; nt < 4; ++nt){
      Frag bb;
      const int nrow = nt*16 + ml;
      bb.q[0] = *(const uint4*)(Bs + nrow*LDT +      khalf*8);
      bb.q[1] = *(const uint4*)(Bs + nrow*LDT + 16 + khalf*8);
      acc[nt] = __builtin_amdgcn_wmma_f32_16x16x32_bf16(
          false, a.v, false, bb.v, (short)0, acc[nt], false, false);
    }
    __syncthreads();
  }
  // C layout: lanes 0-15 N=ml M=r, lanes 16-31 N=ml M=8+r
  #pragma unroll
  for (int nt = 0; nt < 4; ++nt)
    #pragma unroll
    for (int r = 0; r < 8; ++r){
      size_t m = (size_t)(mBase + w*16 + 8*khalf + r);
      logits[m*Ncols + nBlk*64 + nt*16 + ml] = acc[nt][r];
    }
}

// ---------------------------------------------------------------------------
// Kernel 2: per-(b,t,h) softmax groups. trans -> log(0.5*ts+0.5*nn) in place,
// emiss -> log_softmax gathered at obs_idx into lep.
// ---------------------------------------------------------------------------
__global__ __launch_bounds__(256) void postproc(
    float* __restrict__ logits, const float* __restrict__ b_trans,
    const float* __restrict__ b_emiss, const float* __restrict__ unnorm_trans,
    const int* __restrict__ obs_idx, float* __restrict__ lep){
  int g = blockIdx.x*256 + threadIdx.x;
  if (g >= Mrows*32) return;
  int bt = g >> 5, sub = g & 31;
  if (sub < 16){
    int h = sub;
    size_t base = (size_t)bt*Ncols + h*16;
    float x[16], mx = -INFINITY;
    #pragma unroll
    for (int j = 0; j < 16; ++j){ x[j] = logits[base+j] + b_trans[h*16+j]; mx = fmaxf(mx, x[j]); }
    float s = 0.f;
    #pragma unroll
    for (int j = 0; j < 16; ++j) s += expf(x[j]-mx);
    float inv = 1.f/s;
    float t[16], mt = -INFINITY;
    #pragma unroll
    for (int j = 0; j < 16; ++j){ t[j] = unnorm_trans[h*16+j]; mt = fmaxf(mt, t[j]); }
    float st = 0.f;
    #pragma unroll
    for (int j = 0; j < 16; ++j) st += expf(t[j]-mt);
    float invt = 1.f/st;
    #pragma unroll
    for (int j = 0; j < 16; ++j)
      logits[base+j] = logf(0.5f*expf(t[j]-mt)*invt + 0.5f*expf(x[j]-mx)*inv);
  } else {
    int h = sub - 16;
    size_t base = (size_t)bt*Ncols + 256 + h*16;
    float x[16], mx = -INFINITY;
    #pragma unroll
    for (int j = 0; j < 16; ++j){ x[j] = logits[base+j] + b_emiss[h*16+j]; mx = fmaxf(mx, x[j]); }
    float s = 0.f;
    #pragma unroll
    for (int j = 0; j < 16; ++j) s += expf(x[j]-mx);
    float lse = mx + logf(s);
    int o = obs_idx[bt];
    lep[(size_t)bt*16 + h] = x[o] - lse;
  }
}

// 16-lane logsumexp within each half-wave (both halves hold mirrored data)
__device__ __forceinline__ float lse16(float v){
  float m = v;
  #pragma unroll
  for (int o = 1; o < 16; o <<= 1) m = fmaxf(m, __shfl_xor(m, o, 32));
  float s = expf(v - m);
  #pragma unroll
  for (int o = 1; o < 16; o <<= 1) s += __shfl_xor(s, o, 32);
  return m + logf(s);
}

// ---------------------------------------------------------------------------
// Kernel 3: forward/backward recursions. 1 block per b; wave0=alpha, wave1=beta.
// log_trans lives in logits cols [0,256) with row stride Ncols.
// ---------------------------------------------------------------------------
__global__ __launch_bounds__(64) void recur(
    const float* __restrict__ logits, const float* __restrict__ lep,
    const float* __restrict__ sp, float* __restrict__ alpha,
    float* __restrict__ beta){
  const int b = blockIdx.x;
  const int tid = threadIdx.x, wave = tid >> 5, lane = tid & 31, j = lane & 15;
  if (wave == 0){
    float mx = -INFINITY;
    for (int h = 0; h < 16; ++h) mx = fmaxf(mx, sp[h]);
    float s = 0.f;
    for (int h = 0; h < 16; ++h) s += expf(sp[h]-mx);
    float lpj = sp[j] - (mx + logf(s));
    float a = lpj + lep[((size_t)b*Tlen)*16 + j];
    a -= lse16(a);
    if (lane < 16) alpha[((size_t)b*Tlen)*16 + j] = a;
    for (int t = 1; t < Tlen; ++t){
      const float* trp = logits + ((size_t)(b*Tlen + t))*Ncols;
      float tr[16];
      #pragma unroll
      for (int i = 0; i < 16; ++i) tr[i] = trp[i*16 + j];
      float m2 = -INFINITY;
      #pragma unroll
      for (int i = 0; i < 16; ++i) m2 = fmaxf(m2, __shfl(a, i, 32) + tr[i]);
      float s2 = 0.f;
      #pragma unroll
      for (int i = 0; i < 16; ++i) s2 += expf(__shfl(a, i, 32) + tr[i] - m2);
      float anew = lep[((size_t)(b*Tlen + t))*16 + j] + m2 + logf(s2);
      anew -= lse16(anew);
      a = anew;
      if (lane < 16) alpha[((size_t)(b*Tlen + t))*16 + j] = a;
    }
  } else {
    float bc = 0.f;
    if (lane < 16) beta[((size_t)(b*Tlen + Tlen-1))*16 + j] = 0.f;
    for (int t = Tlen-2; t >= 0; --t){
      float v = lep[((size_t)(b*Tlen + t))*16 + j] + bc;   // lep[t][j]+beta[t+1][j]
      const float* trp = logits + ((size_t)(b*Tlen + t))*Ncols + j*16;  // row i=j
      float tr[16];
      #pragma unroll
      for (int jj = 0; jj < 16; ++jj) tr[jj] = trp[jj];
      float m2 = -INFINITY;
      #pragma unroll
      for (int jj = 0; jj < 16; ++jj) m2 = fmaxf(m2, tr[jj] + __shfl(v, jj, 32));
      float s2 = 0.f;
      #pragma unroll
      for (int jj = 0; jj < 16; ++jj) s2 += expf(tr[jj] + __shfl(v, jj, 32) - m2);
      bc = m2 + logf(s2);
      if (lane < 16) beta[((size_t)(b*Tlen + t))*16 + j] = bc;
    }
  }
}

__device__ float blockRedSum(float v){
  __shared__ float sh[8];
  int lane = threadIdx.x & 31, wid = threadIdx.x >> 5;
  #pragma unroll
  for (int o = 16; o; o >>= 1) v += __shfl_xor(v, o, 32);
  __syncthreads();
  if (lane == 0) sh[wid] = v;
  __syncthreads();
  float r;
  if (wid == 0){
    r = (lane < 8) ? sh[lane] : 0.f;
    #pragma unroll
    for (int o = 4; o; o >>= 1) r += __shfl_xor(r, o, 32);
    if (lane == 0) sh[0] = r;
  }
  __syncthreads();
  r = sh[0];
  __syncthreads();
  return r;
}

__device__ float blockRedMax(float v){
  __shared__ float sh[8];
  int lane = threadIdx.x & 31, wid = threadIdx.x >> 5;
  #pragma unroll
  for (int o = 16; o; o >>= 1) v = fmaxf(v, __shfl_xor(v, o, 32));
  __syncthreads();
  if (lane == 0) sh[wid] = v;
  __syncthreads();
  float r;
  if (wid == 0){
    r = (lane < 8) ? sh[lane] : -INFINITY;
    #pragma unroll
    for (int o = 4; o; o >>= 1) r = fmaxf(r, __shfl_xor(r, o, 32));
    if (lane == 0) sh[0] = r;
  }
  __syncthreads();
  r = sh[0];
  __syncthreads();
  return r;
}

// ---------------------------------------------------------------------------
// Kernel 4: per-b loss terms (gamma/emission/prior + xi-weighted transition)
// ---------------------------------------------------------------------------
__global__ __launch_bounds__(256) void loss_k(
    const float* __restrict__ logits, const float* __restrict__ lep,
    const float* __restrict__ alpha, const float* __restrict__ beta,
    const int* __restrict__ seq_lengths, const float* __restrict__ sp,
    float* __restrict__ partials){
  const int b = blockIdx.x, tid = threadIdx.x;
  const int L = seq_lengths[b];
  float local = 0.f;
  // gamma / emission / prior terms (beta rolled by T-L)
  for (int t = tid; t < Tlen; t += 256){
    int rt = (t + Tlen - L) % Tlen;
    float g[16], mx = -INFINITY;
    #pragma unroll
    for (int h = 0; h < 16; ++h){
      g[h] = alpha[((size_t)(b*Tlen + t))*16 + h] + beta[((size_t)(b*Tlen + rt))*16 + h];
      mx = fmaxf(mx, g[h]);
    }
    float s = 0.f;
    #pragma unroll
    for (int h = 0; h < 16; ++h) s += expf(g[h]-mx);
    float z = mx + logf(s);
    if (t < L){
      float e = 0.f;
      #pragma unroll
      for (int h = 0; h < 16; ++h) e += expf(g[h]-z)*lep[((size_t)(b*Tlen + t))*16 + h];
      local += e;
    }
    if (t == 0){
      float m2 = -INFINITY;
      for (int h = 0; h < 16; ++h) m2 = fmaxf(m2, sp[h]);
      float s2 = 0.f;
      for (int h = 0; h < 16; ++h) s2 += expf(sp[h]-m2);
      float lz = m2 + logf(s2);
      float p = 0.f;
      for (int h = 0; h < 16; ++h) p += expf(g[h]-z)*(sp[h]-lz);
      local += p;
    }
  }
  // xi term: thread (i,j) of the 16x16 grid; L uniform per block
  const int i = tid >> 4, jj = tid & 15;
  for (int k = 0; k < Tlen-1 && k < L-1; ++k){
    int rt1 = (k + 1 + Tlen - L) % Tlen;
    float u = logits[((size_t)(b*Tlen + k + 1))*Ncols + i*16 + jj]
            + alpha[((size_t)(b*Tlen + k))*16 + i]
            + lep[((size_t)(b*Tlen + k + 1))*16 + jj]
            + beta[((size_t)(b*Tlen + rt1))*16 + jj];
    float mz   = blockRedMax(u);
    float ssum = blockRedSum(expf(u - mz));
    local += (expf(u - mz)/ssum) * logits[((size_t)(b*Tlen + k))*Ncols + i*16 + jj];
  }
  float tot = blockRedSum(local);
  if (tid == 0) partials[b] = tot;
}

// ---------------------------------------------------------------------------
// Kernel 5: deterministic final mean over b
// ---------------------------------------------------------------------------
__global__ __launch_bounds__(128) void final_k(const float* __restrict__ partials,
                                               float* __restrict__ out){
  __shared__ float sh[4];
  int tid = threadIdx.x, lane = tid & 31, wid = tid >> 5;
  float v = partials[tid];
  #pragma unroll
  for (int o = 16; o; o >>= 1) v += __shfl_xor(v, o, 32);
  if (lane == 0) sh[wid] = v;
  __syncthreads();
  if (tid == 0) out[0] = (sh[0] + sh[1] + sh[2] + sh[3]) / (float)Bsz;
}

extern "C" void kernel_launch(void* const* d_in, const int* in_sizes, int n_in,
                              void* d_out, int out_size, void* d_ws, size_t ws_size,
                              hipStream_t stream) {
  (void)in_sizes; (void)n_in; (void)out_size; (void)ws_size;
  const float* emb          = (const float*)d_in[0];
  const int*   obs_idx      = (const int*)d_in[1];
  const int*   seq_lengths  = (const int*)d_in[2];
  const float* W_trans      = (const float*)d_in[3];
  const float* b_trans      = (const float*)d_in[4];
  const float* W_emiss      = (const float*)d_in[5];
  const float* b_emiss      = (const float*)d_in[6];
  const float* state_priors = (const float*)d_in[7];
  const float* unnorm_trans = (const float*)d_in[8];

  float* ws       = (float*)d_ws;
  float* logits   = ws;                                  // M x 512 (trans becomes log_trans in place)
  float* lep      = logits + (size_t)Mrows*Ncols;        // M x 16
  float* alpha    = lep    + (size_t)Mrows*Hh;           // M x 16
  float* beta     = alpha  + (size_t)Mrows*Hh;           // M x 16
  float* partials = beta   + (size_t)Mrows*Hh;           // B

  gemm_bf16<<<dim3(Mrows/64, 8), 128, 0, stream>>>(emb, W_trans, W_emiss, logits);
  postproc<<<(Mrows*32)/256, 256, 0, stream>>>(logits, b_trans, b_emiss,
                                               unnorm_trans, obs_idx, lep);
  recur<<<Bsz, 64, 0, stream>>>(logits, lep, state_priors, alpha, beta);
  loss_k<<<Bsz, 256, 0, stream>>>(logits, lep, alpha, beta, seq_lengths,
                                  state_priors, partials);
  final_k<<<1, 128, 0, stream>>>(partials, (float*)d_out);
}
